// Head_70892730187993
// MI455X (gfx1250) — compile-verified, matrix operand order checked by reading, not run on hardware
//
#include <hip/hip_runtime.h>
#include <hip/hip_bf16.h>

typedef __attribute__((ext_vector_type(16))) _Float16 v16h;
typedef __attribute__((ext_vector_type(8)))  _Float16 v8h;
typedef __attribute__((ext_vector_type(8)))  float    v8f;
typedef __attribute__((ext_vector_type(4)))  float    v4f;
typedef __attribute__((ext_vector_type(4)))  unsigned v4u;
typedef __attribute__((ext_vector_type(8)))  int      v8i;
typedef __attribute__((ext_vector_type(4)))  int      v4i;

#define T_SEQ 2048
#define C_DIM 1024
#define H_DIM 128
#define B_DIM 8

#define KSTRIDE 136   // K tile row stride in halves (272 B = 256 B data + 16 B pad)
#define VSTRIDE 72    // V tile row stride in halves (144 B = 128 B data + 16 B pad)

#if __has_builtin(__builtin_amdgcn_tensor_load_to_lds)
#define HAVE_TDM 1
#else
#define HAVE_TDM 0
#endif

// Load a 16-half A/B fragment for this lane from a row pointer already offset
// by grp*8 halves.  A-layout: e0..7 <-> k = grp*8+e ; e8..15 <-> k = 16+grp*8+e
__device__ __forceinline__ v16h load_frag16(const _Float16* p) {
    v8h lo = *(const v8h*)(p);
    v8h hi = *(const v8h*)(p + 16);
    v16h r;
#pragma unroll
    for (int i = 0; i < 8; ++i) { r[i] = lo[i]; r[i + 8] = hi[i]; }
    return r;
}

// Workgroup-relative LDS byte offset (low 32 bits of generic address, ISA §10.2).
__device__ __forceinline__ unsigned lds_off(const void* p) {
    return (unsigned)(size_t)p;
}

// CDNA5 async copy global -> LDS (ASYNCcnt), ISA §15.18.3 op 98.
__device__ __forceinline__ void async_ld_b128(unsigned dst_lds, const void* src) {
    asm volatile("global_load_async_to_lds_b128 %0, %1, off"
                 :: "v"(dst_lds), "v"(src) : "memory");
}
__device__ __forceinline__ void wait_async0() {
    asm volatile("s_wait_asynccnt 0x0" ::: "memory");
}

#if HAVE_TDM
// One-shot 2D TDM tile load with LDS row padding (ISA §8 D# layout).
// data_size = 2B elements; dims/strides in elements; pad codes per group1 spec.
// This toolchain exposes the 6-arg builtin: (v4u, v8i, v4i, v4i, v8i, i32).
__device__ __forceinline__ void tdm_load_2d(
    unsigned lds_addr, const void* gaddr,
    unsigned tensor_d0, unsigned tensor_d1,
    unsigned tile_d0, unsigned tile_d1,
    unsigned d0_stride, unsigned pad_interval_code, unsigned pad_amount_code)
{
    const unsigned long long ga = (unsigned long long)gaddr;
    v4u g0;
    g0[0] = 1u;                                        // count=1, user descriptor
    g0[1] = lds_addr;                                  // lds_addr [63:32]
    g0[2] = (unsigned)(ga & 0xffffffffu);              // global_addr [95:64]
    g0[3] = (unsigned)((ga >> 32) & 0x01ffffffu)       // global_addr [120:96]
          | (2u << 30);                                // type=2 ("image")
    v8i g1;
    g1[0] = (int)((1u << 16)                           // data_size = 2 bytes
                | (1u << 20)                           // pad_enable
                | (pad_interval_code << 22)
                | (pad_amount_code << 25));
    g1[1] = (int)((tensor_d0 & 0xffffu) << 16);        // tensor_dim0[15:0]
    g1[2] = (int)((tensor_d0 >> 16) | ((tensor_d1 & 0xffffu) << 16));
    g1[3] = (int)((tensor_d1 >> 16) | (tile_d0 << 16));// tile_dim0
    g1[4] = (int)(tile_d1 & 0xffffu);                  // tile_dim1, tile_dim2=0
    g1[5] = (int)d0_stride;                            // tensor_dim0_stride[31:0]
    g1[6] = 0;                                         // stride hi, dim1_stride lo
    g1[7] = 0;
    v4i g2; g2[0] = 1; g2[1] = 0; g2[2] = 0; g2[3] = 0;// tensor_dim2=1, rest 0
    v4i g3; g3[0] = 0; g3[1] = 0; g3[2] = 0; g3[3] = 0;
    v8i g4;
#pragma unroll
    for (int i = 0; i < 8; ++i) g4[i] = 0;             // extra group: zero-filled
    __builtin_amdgcn_tensor_load_to_lds(g0, g1, g2, g3, g4, 0);
}
#endif

// ---------------------------------------------------------------------------
// Kernel 1: fused QKV projection.  [16384 x 1024] x [1024 x 128] per matrix.
// ---------------------------------------------------------------------------
__global__ __launch_bounds__(128) void qkv_proj_kernel(
    const float* __restrict__ x,
    const float* __restrict__ Wq, const float* __restrict__ Wk,
    const float* __restrict__ Wv,
    _Float16* __restrict__ qo, _Float16* __restrict__ ko,
    _Float16* __restrict__ vto)
{
    const int lane = threadIdx.x & 31;
    const int wave = threadIdx.x >> 5;
    const int l16  = lane & 15;
    const int grp  = lane >> 4;

    const int w   = blockIdx.x * 4 + wave;
    const int mat = w >> 11;
    const int rem = w & 2047;
    const int rowbase = (rem >> 3) * 64;
    const int nbase   = (rem & 7) * 16;

    const float* W = (mat == 0) ? Wq : (mat == 1) ? Wk : Wv;

    v8f acc[4];
#pragma unroll
    for (int t = 0; t < 4; ++t)
#pragma unroll
        for (int r = 0; r < 8; ++r) acc[t][r] = 0.0f;

    for (int kk = 0; kk < C_DIM; kk += 32) {
        v16h bf;
#pragma unroll
        for (int e = 0; e < 16; ++e) {
            const int k = ((e < 8) ? 0 : 16) + grp * 8 + (e & 7);
            bf[e] = (_Float16)W[(size_t)(kk + k) * H_DIM + nbase + l16];
        }
#pragma unroll
        for (int t = 0; t < 4; ++t) {
            const float* xr =
                x + (size_t)(rowbase + t * 16 + l16) * C_DIM + kk + grp * 8;
            v4f a0 = *(const v4f*)(xr);
            v4f a1 = *(const v4f*)(xr + 4);
            v4f a2 = *(const v4f*)(xr + 16);
            v4f a3 = *(const v4f*)(xr + 20);
            v16h af;
#pragma unroll
            for (int i = 0; i < 4; ++i) {
                af[i]      = (_Float16)a0[i];
                af[4 + i]  = (_Float16)a1[i];
                af[8 + i]  = (_Float16)a2[i];
                af[12 + i] = (_Float16)a3[i];
            }
            acc[t] = __builtin_amdgcn_wmma_f32_16x16x32_f16(
                false, af, false, bf, (short)0, acc[t], false, false);
        }
    }

    const float scale = (mat == 0) ? 0.03125f : 1.0f;
#pragma unroll
    for (int t = 0; t < 4; ++t) {
#pragma unroll
        for (int r = 0; r < 8; ++r) {
            const int row = rowbase + t * 16 + grp * 8 + r;
            const int col = nbase + l16;
            const _Float16 vv = (_Float16)(acc[t][r] * scale);
            if (mat == 0) {
                qo[(size_t)row * H_DIM + col] = vv;
            } else if (mat == 1) {
                ko[(size_t)row * H_DIM + col] = vv;
            } else {
                const int bb = row >> 11, tt = row & (T_SEQ - 1);
                vto[((size_t)bb * H_DIM + col) * T_SEQ + tt] = vv;
            }
        }
    }
}

// ---------------------------------------------------------------------------
// Kernel 2: fused causal flash attention.
// K tiles staged per-block via TDM (tensor_load_to_lds, TENSORcnt, LDS row
// padding in-unit), V tiles via async copies (ASYNCcnt) — two independent
// DMA streams double-buffered against the WMMA compute.
// ---------------------------------------------------------------------------
__global__ __launch_bounds__(128) void flash_attn_kernel(
    const _Float16* __restrict__ q, const _Float16* __restrict__ k,
    const _Float16* __restrict__ vt, float* __restrict__ out)
{
    __shared__ __align__(16) _Float16 kbuf[2][64][KSTRIDE];
    __shared__ __align__(16) _Float16 vbuf[2][H_DIM][VSTRIDE];
    __shared__ __align__(16) _Float16 pbuf[4][16][80];

    const int tid  = threadIdx.x;
    const int lane = tid & 31;
    const int wave = tid >> 5;
    const int l16  = lane & 15;
    const int grp  = lane >> 4;

    const int b      = blockIdx.x >> 5;
    const int qblk   = blockIdx.x & 31;
    const int q_base = qblk * 64 + wave * 16;

    const _Float16* Qb = q  + (size_t)b * T_SEQ * H_DIM;
    const _Float16* Kb = k  + (size_t)b * T_SEQ * H_DIM;
    const _Float16* Vb = vt + (size_t)b * H_DIM * T_SEQ;

    v16h qf[4];
#pragma unroll
    for (int c = 0; c < 4; ++c)
        qf[c] = load_frag16(Qb + (size_t)(q_base + l16) * H_DIM + c * 32 + grp * 8);

    v8f o[8];
#pragma unroll
    for (int ht = 0; ht < 8; ++ht)
#pragma unroll
        for (int r = 0; r < 8; ++r) o[ht][r] = 0.0f;

    float m[8], l[8];
#pragma unroll
    for (int r = 0; r < 8; ++r) { m[r] = -__builtin_inff(); l[r] = 0.0f; }

    // ---- stage K (TDM, wave 0 only) + V (async, all threads) for block kb ----
    auto stage_tiles = [&](int kb, int bufi) {
#if HAVE_TDM
        if (wave == 0) {
            // 64 rows x 128 halves, global stride 128 halves; pad 4 DWORDs
            // after every 64 DWORDs (256B row -> 272B LDS stride).
            tdm_load_2d(lds_off(&kbuf[bufi][0][0]), Kb + (size_t)kb * H_DIM,
                        H_DIM, T_SEQ, H_DIM, 64, H_DIM, 5, 3);
        }
#else
#pragma unroll
        for (int i = 0; i < 8; ++i) {
            const int chunk = i * 128 + tid;
            const int row = chunk >> 4;
            const int co  = (chunk & 15) * 8;
            async_ld_b128(lds_off(&kbuf[bufi][row][co]),
                          Kb + (size_t)(kb + row) * H_DIM + co);
        }
#endif
#pragma unroll
        for (int i = 0; i < 8; ++i) {            // V^T: 128 x 64 halves
            const int chunk = i * 128 + tid;
            const int row = chunk >> 3;
            const int co  = (chunk & 7) * 8;
            async_ld_b128(lds_off(&vbuf[bufi][row][co]),
                          Vb + (size_t)row * T_SEQ + kb + co);
        }
    };
    auto wait_stage = [&]() {
#if HAVE_TDM
        if (wave == 0) __builtin_amdgcn_s_wait_tensorcnt(0);
#endif
        wait_async0();
        __syncthreads();
    };

    const int nkb = qblk + 1;        // identical trip count for all 4 waves
    stage_tiles(0, 0);
    wait_stage();

    for (int it = 0; it < nkb; ++it) {
        const int kb  = it * 64;
        const int cur = it & 1;
        if (it + 1 < nkb) stage_tiles(kb + 64, cur ^ 1);

        // ---- S = Q K^T from LDS K tile ----
        v8f s[4];
#pragma unroll
        for (int nt = 0; nt < 4; ++nt) {
#pragma unroll
            for (int r = 0; r < 8; ++r) s[nt][r] = 0.0f;
            const _Float16* kp = &kbuf[cur][nt * 16 + l16][grp * 8];
#pragma unroll
            for (int c = 0; c < 4; ++c) {
                v16h bf = load_frag16(kp + c * 32);
                s[nt] = __builtin_amdgcn_wmma_f32_16x16x32_f16(
                    false, qf[c], false, bf, (short)0, s[nt], false, false);
            }
        }

        // ---- causal mask on diagonal block ----
        if (kb + 63 > q_base) {
#pragma unroll
            for (int nt = 0; nt < 4; ++nt) {
                const int kc = kb + nt * 16 + l16;
#pragma unroll
                for (int r = 0; r < 8; ++r) {
                    const int qr = q_base + grp * 8 + r;
                    if (kc > qr) s[nt][r] = -__builtin_inff();
                }
            }
        }

        // ---- online softmax ----
        float curm[8];
#pragma unroll
        for (int r = 0; r < 8; ++r)
            curm[r] = fmaxf(fmaxf(s[0][r], s[1][r]), fmaxf(s[2][r], s[3][r]));
#pragma unroll
        for (int mk = 1; mk < 16; mk <<= 1)
#pragma unroll
            for (int r = 0; r < 8; ++r)
                curm[r] = fmaxf(curm[r], __shfl_xor(curm[r], mk, 32));

        float al[8];
#pragma unroll
        for (int r = 0; r < 8; ++r) {
            const float nm = fmaxf(m[r], curm[r]);
            al[r] = __expf(m[r] - nm);
            m[r]  = nm;
        }

        float rs[8];
#pragma unroll
        for (int r = 0; r < 8; ++r) rs[r] = 0.0f;
#pragma unroll
        for (int nt = 0; nt < 4; ++nt)
#pragma unroll
            for (int r = 0; r < 8; ++r) {
                const float pv = __expf(s[nt][r] - m[r]);
                s[nt][r] = pv;
                rs[r] += pv;
            }
#pragma unroll
        for (int mk = 1; mk < 16; mk <<= 1)
#pragma unroll
            for (int r = 0; r < 8; ++r)
                rs[r] += __shfl_xor(rs[r], mk, 32);
#pragma unroll
        for (int r = 0; r < 8; ++r) l[r] = l[r] * al[r] + rs[r];

        // ---- P (C/D layout) -> per-wave LDS -> A-layout fragments ----
#pragma unroll
        for (int nt = 0; nt < 4; ++nt)
#pragma unroll
            for (int r = 0; r < 8; ++r)
                pbuf[wave][grp * 8 + r][nt * 16 + l16] = (_Float16)s[nt][r];

#pragma unroll
        for (int ht = 0; ht < 8; ++ht)
#pragma unroll
            for (int r = 0; r < 8; ++r) o[ht][r] *= al[r];

        v16h pa[2];
#pragma unroll
        for (int c = 0; c < 2; ++c)
            pa[c] = load_frag16(&pbuf[wave][l16][c * 32 + grp * 8]);

        // ---- O += P V from LDS V tile ----
#pragma unroll
        for (int ht = 0; ht < 8; ++ht) {
            const _Float16* vp = &vbuf[cur][ht * 16 + l16][grp * 8];
#pragma unroll
            for (int c = 0; c < 2; ++c) {
                v16h bf = load_frag16(vp + c * 32);
                o[ht] = __builtin_amdgcn_wmma_f32_16x16x32_f16(
                    false, pa[c], false, bf, (short)0, o[ht], false, false);
            }
        }

        wait_stage();        // prefetched tiles resident, cur safe to overwrite
    }

    // ---- epilogue: normalize and store fp32 ----
#pragma unroll
    for (int ht = 0; ht < 8; ++ht) {
#pragma unroll
        for (int r = 0; r < 8; ++r) {
            const int qr = q_base + grp * 8 + r;
            out[((size_t)b * T_SEQ + qr) * H_DIM + ht * 16 + l16] =
                o[ht][r] / l[r];
        }
    }
}

extern "C" void kernel_launch(void* const* d_in, const int* in_sizes, int n_in,
                              void* d_out, int out_size, void* d_ws,
                              size_t ws_size, hipStream_t stream) {
    const float* x  = (const float*)d_in[0];
    const float* Wq = (const float*)d_in[1];
    const float* Wk = (const float*)d_in[2];
    const float* Wv = (const float*)d_in[3];
    float* out = (float*)d_out;

    _Float16* qf  = (_Float16*)d_ws;
    _Float16* kf  = qf + (size_t)B_DIM * T_SEQ * H_DIM;
    _Float16* vtf = kf + (size_t)B_DIM * T_SEQ * H_DIM;

    qkv_proj_kernel<<<1536, 128, 0, stream>>>(x, Wq, Wk, Wv, qf, kf, vtf);
    flash_attn_kernel<<<256, 128, 0, stream>>>(qf, kf, vtf, out);
}
